// EdgeConv_27410481283234
// MI455X (gfx1250) — matrix operand order repaired and checked by reading, not compiled, not run on hardware
//
#include <hip/hip_runtime.h>

typedef __attribute__((ext_vector_type(16))) _Float16 v16h;
typedef __attribute__((ext_vector_type(8)))  _Float16 v8h;
typedef __attribute__((ext_vector_type(8)))  float    v8f;

#define BB    2
#define NN    8192
#define CC    64
#define KK    20
#define OUTC  64
#define ED    197
#define KP    224          // padded feature dim: 7 k-steps of 32
#define KS    7            // k-steps
#define PTS   4            // points per workgroup
#define ROWS  (PTS*KK)     // 80 edge rows per workgroup
#define MT    5            // M tiles (80/16)

// ---------------------------------------------------------------------------
// Kernel 0: pack conv_w (f32, [64][197]) into WMMA B-fragment layout (f16).
// B matrix is Kdim x N. This code uses contiguous k order (same as its A): lane l holds N = l&15,
// element e holds K = e + (l<16 ? 0 : 16) within a 32-deep k-step.
// Fragment (nt, ks) stored at [(nt*7+ks)*32 + lane]*16 halves (32B/lane).
// ---------------------------------------------------------------------------
__global__ void packW(const float* __restrict__ w, _Float16* __restrict__ wp) {
    int frag = blockIdx.x;            // 0..27 = nt*KS + ks
    int lane = threadIdx.x;           // 0..31
    int nt = frag / KS, ks = frag % KS;
    int n    = lane & 15;
    int koff = (lane & 16) ? 16 : 0;
    v16h vals;
#pragma unroll
    for (int h = 0; h < 16; ++h) {
        int kk = ks * 32 + h + koff;
        float v = (kk < ED) ? w[(nt * 16 + n) * ED + kk] : 0.0f;
        vals[h] = (_Float16)v;
    }
    *reinterpret_cast<v16h*>(wp + ((size_t)frag * 32 + lane) * 16) = vals;
}

// ---------------------------------------------------------------------------
// Kernel 1: exact KNN (top-20 smallest squared distance, self included —
// matches top_k of -dist^2 where pd[i][i]=0 is max). One thread per point,
// candidate xyz tiled through LDS, register-resident top-20.
// ---------------------------------------------------------------------------
__global__ __launch_bounds__(256)
void knn(const float* __restrict__ x, unsigned short* __restrict__ outIdx) {
    __shared__ float sT[3][256];
    const int b = blockIdx.y;
    const int i = blockIdx.x * 256 + threadIdx.x;
    const float* xb = x + (size_t)b * CC * NN;

    const float xi0 = xb[i];
    const float xi1 = xb[NN + i];
    const float xi2 = xb[2 * NN + i];

    float bD[KK];
    int   bI[KK];
#pragma unroll
    for (int q = 0; q < KK; ++q) { bD[q] = 3.4e38f; bI[q] = i; }
    float worst = 3.4e38f;

    for (int t0 = 0; t0 < NN; t0 += 256) {
        __syncthreads();
        sT[0][threadIdx.x] = xb[t0 + threadIdx.x];
        sT[1][threadIdx.x] = xb[NN + t0 + threadIdx.x];
        sT[2][threadIdx.x] = xb[2 * NN + t0 + threadIdx.x];
        __syncthreads();
        for (int jj = 0; jj < 256; ++jj) {
            float dx = xi0 - sT[0][jj];
            float dy = xi1 - sT[1][jj];
            float dz = xi2 - sT[2][jj];
            float d2 = dx * dx + dy * dy + dz * dz;
            if (d2 < worst) {
                int j = t0 + jj;
                bool done = false;
#pragma unroll
                for (int q = 0; q < KK; ++q) {
                    if (!done && bD[q] == worst) { bD[q] = d2; bI[q] = j; done = true; }
                }
                worst = bD[0];
#pragma unroll
                for (int q = 1; q < KK; ++q) worst = fmaxf(worst, bD[q]);
            }
        }
    }
    unsigned short* dst = outIdx + ((size_t)b * NN + i) * KK;
#pragma unroll
    for (int q = 0; q < KK; ++q) dst[q] = (unsigned short)bI[q];
}

// ---------------------------------------------------------------------------
// Kernel 2: fused gather + 197-dim edge features + WMMA GEMM (f16 in / f32
// acc) + batchnorm + leaky-relu + max over k. 128 threads = 4 waves; wave w
// owns output-channel tile w; 80 rows = 5 M-tiles; 7 k-steps of 32.
// ---------------------------------------------------------------------------
__global__ __launch_bounds__(128)
void edgeconv(const float* __restrict__ x, const _Float16* __restrict__ wp,
              const unsigned short* __restrict__ nbr,
              const float* __restrict__ gamma, const float* __restrict__ beta,
              const float* __restrict__ mean,  const float* __restrict__ var,
              float* __restrict__ out) {
    __shared__ _Float16 sE[ROWS * KP];       // 35840 B edge matrix (f16)
    __shared__ float    sY[ROWS * OUTC];     // 20480 B pre-max activations
    __shared__ float    sCtr[PTS * CC];      // center features
    __shared__ float    sVec[ROWS * 3];      // neighbor - center xyz
    __shared__ float    sDist[ROWS];
    __shared__ int      sIdx[ROWS];
    __shared__ float    sMeanD[PTS], sVarD[PTS], sRelH[PTS], sMd[PTS][3];

    const int tid  = threadIdx.x;
    const int b    = blockIdx.y;
    const int base = blockIdx.x * PTS;
    const float* xb = x + (size_t)b * CC * NN;
    const unsigned short* idxB = nbr + ((size_t)b * NN + base) * KK;

    // ---- phase 1: neighbor indices + center features -----------------------
    if (tid < ROWS) sIdx[tid] = (int)idxB[tid];
#pragma unroll
    for (int s = 0; s < 2; ++s) {
        int o = tid + s * 128;               // 0..255 -> (p, c)
        int p = o >> 6, c = o & 63;
        sCtr[o] = xb[(size_t)c * NN + (base + p)];
    }
    __syncthreads();

    // ---- phase 2: base features [ctr | nb-ctr | nb] + vec/dist -------------
    for (int o = tid; o < ROWS * CC; o += 128) {
        int r = o >> 6, c = o & 63;
        int j = sIdx[r];
        float ctr = sCtr[(r / KK) * CC + c];
        float nb  = xb[(size_t)c * NN + j];
        sE[r * KP + c]       = (_Float16)ctr;
        sE[r * KP + 64 + c]  = (_Float16)(nb - ctr);
        sE[r * KP + 128 + c] = (_Float16)nb;
    }
    if (tid < ROWS) {
        int p = tid / KK;
        int j = sIdx[tid];
        float vx = xb[j]          - sCtr[p * CC + 0];
        float vy = xb[NN + j]     - sCtr[p * CC + 1];
        float vz = xb[2 * NN + j] - sCtr[p * CC + 2];
        sVec[tid * 3 + 0] = vx; sVec[tid * 3 + 1] = vy; sVec[tid * 3 + 2] = vz;
        sDist[tid] = sqrtf(vx * vx + vy * vy + vz * vz);
    }
    __syncthreads();

    // ---- phase 3: per-point stats (mean dist, var ddof=1, rel_h, mean dir) -
    if (tid < PTS) {
        float sd = 0.f, sd2 = 0.f, mz = 0.f, m0 = 0.f, m1 = 0.f, m2 = 0.f;
#pragma unroll
        for (int q = 0; q < KK; ++q) {
            int r = tid * KK + q;
            float d = sDist[r];
            sd += d; sd2 += d * d;
            mz += sVec[r * 3 + 2];
            float inv = 1.0f / (d + 1e-6f);
            m0 += sVec[r * 3 + 0] * inv;
            m1 += sVec[r * 3 + 1] * inv;
            m2 += sVec[r * 3 + 2] * inv;
        }
        float meanD = sd * (1.0f / KK);
        sMeanD[tid] = meanD;
        sVarD[tid]  = (sd2 - (float)KK * meanD * meanD) * (1.0f / (KK - 1));
        sRelH[tid]  = -mz * (1.0f / KK);     // z_ctr - mean(z_nb) = -mean(vec_z)
        m0 *= (1.0f / KK); m1 *= (1.0f / KK); m2 *= (1.0f / KK);
        float nrm = sqrtf(m0 * m0 + m1 * m1 + m2 * m2) + 1e-6f;
        sMd[tid][0] = m0 / nrm; sMd[tid][1] = m1 / nrm; sMd[tid][2] = m2 / nrm;
    }
    __syncthreads();

    // ---- phase 4: tail features (dist, density, rel_h, var, angle) + pad ---
    if (tid < ROWS) {
        int p = tid / KK;
        float d   = sDist[tid];
        float inv = 1.0f / (d + 1e-6f);
        float cs  = sVec[tid * 3 + 0] * inv * sMd[p][0]
                  + sVec[tid * 3 + 1] * inv * sMd[p][1]
                  + sVec[tid * 3 + 2] * inv * sMd[p][2];
        _Float16* row = sE + tid * KP;
        row[192] = (_Float16)d;
        row[193] = (_Float16)sMeanD[p];
        row[194] = (_Float16)sRelH[p];
        row[195] = (_Float16)sVarD[p];
        row[196] = (_Float16)(1.0f - fabsf(cs));
#pragma unroll
        for (int c = ED; c < KP; ++c) row[c] = (_Float16)0.0f;
    }
    __syncthreads();

    // ---- phase 5: WMMA GEMM + BN + leaky relu ------------------------------
    const int lane = tid & 31;
    const int nt   = tid >> 5;               // wave id == output-channel tile
    v16h bf[KS];
#pragma unroll
    for (int ks = 0; ks < KS; ++ks)
        bf[ks] = *reinterpret_cast<const v16h*>(wp + ((size_t)(nt * KS + ks) * 32 + lane) * 16);

    const int mrow = lane & 15;
    const int koff = (lane & 16) ? 16 : 0;   // A layout: lanes>=16 hold K 16-31
    const int col  = nt * 16 + mrow;
    float scale = gamma[col] * rsqrtf(var[col] + 1e-5f);
    float shift = beta[col] - mean[col] * scale;

    for (int mt = 0; mt < MT; ++mt) {
        v8f acc = {};
        const _Float16* rowp = sE + (size_t)(mt * 16 + mrow) * KP;
#pragma unroll
        for (int ks = 0; ks < KS; ++ks) {
            v8h lo = *reinterpret_cast<const v8h*>(rowp + ks * 32 + koff);
            v8h hi = *reinterpret_cast<const v8h*>(rowp + ks * 32 + 8 + koff);
            v16h a;
#pragma unroll
            for (int h = 0; h < 8; ++h) { a[h] = lo[h]; a[8 + h] = hi[h]; }
            acc = __builtin_amdgcn_wmma_f32_16x16x32_f16(
                      false, a, false, bf[ks], (short)0, acc, false, false);
        }
#pragma unroll
        for (int v = 0; v < 8; ++v) {
            int r = mt * 16 + ((lane & 16) ? (v + 8) : v);
            float y = acc[v] * scale + shift;
            y = (y >= 0.0f) ? y : 0.2f * y;
            sY[r * OUTC + col] = y;
        }
    }
    __syncthreads();

    // ---- phase 6: max over k, write (B, OUTC, N) ---------------------------
#pragma unroll
    for (int s = 0; s < 2; ++s) {
        int o = tid + s * 128;               // 0..255 -> (p, c)
        int p = o >> 6, c = o & 63;
        float m = sY[(p * KK) * OUTC + c];
#pragma unroll
        for (int q = 1; q < KK; ++q) m = fmaxf(m, sY[(p * KK + q) * OUTC + c]);
        out[((size_t)b * OUTC + c) * NN + (base + p)] = m;
    }
}

// ---------------------------------------------------------------------------
extern "C" void kernel_launch(void* const* d_in, const int* in_sizes, int n_in,
                              void* d_out, int out_size, void* d_ws, size_t ws_size,
                              hipStream_t stream) {
    const float* x       = (const float*)d_in[0];
    const float* conv_w  = (const float*)d_in[1];
    const float* bn_g    = (const float*)d_in[2];
    const float* bn_b    = (const float*)d_in[3];
    const float* bn_m    = (const float*)d_in[4];
    const float* bn_v    = (const float*)d_in[5];
    float* out           = (float*)d_out;

    _Float16*       wsW   = (_Float16*)d_ws;                       // 28672 B
    unsigned short* wsIdx = (unsigned short*)((char*)d_ws + 32768); // 655360 B

    packW<<<dim3(4 * KS), dim3(32), 0, stream>>>(conv_w, wsW);
    knn<<<dim3(NN / 256, BB), dim3(256), 0, stream>>>(x, wsIdx);
    edgeconv<<<dim3(NN / PTS, BB), dim3(128), 0, stream>>>(
        x, wsW, wsIdx, bn_g, bn_b, bn_m, bn_v, out);
}
